// Correlation_76639396430164
// MI455X (gfx1250) — matrix operand order repaired and checked by reading, not compiled, not run on hardware
//
#include <hip/hip_runtime.h>

typedef __attribute__((ext_vector_type(16))) __bf16 bf16x16;
typedef __attribute__((ext_vector_type(8)))  __bf16 bf16x8;
typedef __attribute__((ext_vector_type(8)))  float  f32x8;
typedef __attribute__((ext_vector_type(4)))  float  f32x4;

namespace {
constexpr int PATCH = 21;          // patch size (displacements per axis)
constexpr int NB = 8, NC = 256, NH = 96, NW = 128;
constexpr int CCH    = 128;        // channels resident in LDS per chunk
constexpr int NCHUNK = NC / CCH;   // 2
constexpr int RS     = CCH + 8;    // LDS row stride in bf16 (136 -> 272B rows, 16B aligned, bank-friendly)
constexpr int PLANE  = NW * RS;    // 128 pixel rows (64 even-parity + 64 odd-parity)
}

// Build a 16x16x32 bf16 WMMA fragment from two 16B LDS reads (ds_load_b128 x2).
static __device__ __forceinline__ bf16x16 ld_frag(const __bf16* p0, const __bf16* p1) {
  bf16x8 a = *(const bf16x8*)p0;
  bf16x8 b = *(const bf16x8*)p1;
  return __builtin_shufflevector(a, b, 0,1,2,3,4,5,6,7,8,9,10,11,12,13,14,15);
}

__global__ __launch_bounds__(128)
void corr_wmma_kernel(const float* __restrict__ in1,
                      const float* __restrict__ in2,
                      float* __restrict__ out) {
  // LDS: hi/lo bf16 planes for the X1 row and the shifted X2 row, parity-grouped
  // (rows 0..63 = even pixels, 64..127 = odd pixels), plus the 21x128 output slab.
  // Total = 4*34816 + 10752 = 150016 B (CDNA5 allows up to 320KB/WG).
  __shared__ __align__(16) __bf16 sX1h[PLANE];
  __shared__ __align__(16) __bf16 sX1l[PLANE];
  __shared__ __align__(16) __bf16 sX2h[PLANE];
  __shared__ __align__(16) __bf16 sX2l[PLANE];
  __shared__ __align__(16) float  sOut[PATCH * NW];

  const int t     = threadIdx.x;
  const int lane  = t & 31;
  const int wv    = t >> 5;        // wave id 0..3
  const int nlo   = lane & 15;     // row/col within tile this lane owns
  const int hi    = lane >> 4;     // lane-group (selects K sub-ranges / M+8)

  int bid = blockIdx.x;
  const int iy = bid % PATCH;  bid /= PATCH;
  const int h  = bid % NH;
  const int b  = bid / NH;
  const int h2 = h + 2 * iy - 20;                 // shifted input2 row (may be OOB)

  // zero the output slab (also the final answer when h2 is out of range)
  for (int r = 0; r < PATCH; ++r) sOut[r * NW + t] = 0.0f;

  const int par   = wv >> 1;       // parity handled by this wave (0: even w, 1: odd w)
  const int halfv = wv & 1;        // which 32-row half of the 64-row parity GEMM

  // 2 row-tiles x 3 col-tiles of the banded Gram matrix, f32 accumulators
  f32x8 acc[2][3] = {};

  if (0 <= h2 && h2 < NH) {
    for (int cc = 0; cc < NCHUNK; ++cc) {
      const int c0 = cc * CCH;
      __syncthreads();   // protect LDS planes from the previous chunk's readers

      // ---- stage X1 (row h) and X2 (row h2): fp32 -> bf16 hi + bf16 lo ----
      {
        const int q  = t & 31;     // float4 index along W (w = 4q..4q+3)
        const int cb = t >> 5;     // channel sub-phase 0..3
        for (int s = 0; s < 2; ++s) {
          const float* src = s ? in2 : in1;
          const int   hrow = s ? h2 : h;
          __bf16* dh = s ? sX2h : sX1h;
          __bf16* dl = s ? sX2l : sX1l;
          for (int k = 0; k < CCH / 4; ++k) {
            const int c = c0 + cb + 4 * k;
            f32x4 v = *(const f32x4*)(src + ((size_t)(b * NC + c) * NH + hrow) * NW + 4 * q);
#pragma unroll
            for (int r = 0; r < 4; ++r) {
              // pixel w = 4q + r -> parity (r&1), parity-index i = 2q + (r>>1)
              const int row = (r & 1) * 64 + 2 * q + (r >> 1);
              float  x  = v[r];
              __bf16 xh = (__bf16)x;
              __bf16 xl = (__bf16)(x - (float)xh);
              dh[row * RS + (c - c0)] = xh;
              dl[row * RS + (c - c0)] = xl;
            }
          }
        }
      }
      __syncthreads();

      // ---- banded GEMM over this channel chunk via bf16 WMMA (bf16x3 split) ----
      for (int kc = 0; kc < CCH / 32; ++kc) {
        const int k0 = kc * 32;

        // A fragments (16x32): lane-group 0 holds K {0..7,16..23}, group 1 {8..15,24..31}
        bf16x16 Ah[2], Al[2];
#pragma unroll
        for (int rt = 0; rt < 2; ++rt) {
          const int row = par * 64 + 16 * (2 * halfv + rt) + nlo;
          const __bf16* bh = sX1h + row * RS + k0 + hi * 8;
          const __bf16* bl = sX1l + row * RS + k0 + hi * 8;
          Ah[rt] = ld_frag(bh, bh + 16);
          Al[rt] = ld_frag(bl, bl + 16);
        }

#pragma unroll
        for (int ct = 0; ct < 3; ++ct) {
          // B fragments (32x16): lane n = column; group 0 K {0..7,16..23}, group 1 K {8..15,24..31}
          const int row = par * 64 + 16 * (halfv + ct) + nlo;
          const __bf16* bh = sX2h + row * RS + k0 + hi * 8;
          const __bf16* bl = sX2l + row * RS + k0 + hi * 8;
          bf16x16 Bh = ld_frag(bh, bh + 16);
          bf16x16 Bl = ld_frag(bl, bl + 16);
#pragma unroll
          for (int rt = 0; rt < 2; ++rt) {
            acc[rt][ct] = __builtin_amdgcn_wmma_f32_16x16x32_bf16(
                false, Ah[rt], false, Bh, (short)0, acc[rt][ct], false, false);
            acc[rt][ct] = __builtin_amdgcn_wmma_f32_16x16x32_bf16(
                false, Ah[rt], false, Bl, (short)0, acc[rt][ct], false, false);
            acc[rt][ct] = __builtin_amdgcn_wmma_f32_16x16x32_bf16(
                false, Al[rt], false, Bh, (short)0, acc[rt][ct], false, false);
          }
        }
      }
    }

    // ---- scatter band diagonals into the slab: G[i,j] -> out[ix = j-i+10][w = 2i+par] ----
#pragma unroll
    for (int rt = 0; rt < 2; ++rt) {
#pragma unroll
      for (int ct = 0; ct < 3; ++ct) {
        const int i0 = 16 * (2 * halfv + rt);
        const int j0 = 16 * (halfv + ct);
#pragma unroll
        for (int r = 0; r < 8; ++r) {
          const int m  = r + 8 * hi;                 // C/D layout: VGPR r, lanes 16-31 -> M+8
          const int ix = (j0 + nlo) - (i0 + m) + 10; // displacement index
          if (0 <= ix && ix < PATCH)
            sOut[ix * NW + 2 * (i0 + m) + par] = acc[rt][ct][r];
        }
      }
    }
  }
  __syncthreads();

  // ---- coalesced store: normalize by C, leaky ReLU(0.1) ----
  const size_t obase = ((size_t)(b * (PATCH * PATCH) + iy * PATCH) * NH + h) * NW;
  for (int ix = wv; ix < PATCH; ix += 4) {
    f32x4 v = *(const f32x4*)(sOut + ix * NW + 4 * lane);
#pragma unroll
    for (int r = 0; r < 4; ++r) {
      float y = v[r] * (1.0f / NC);
      v[r] = (y >= 0.0f) ? y : 0.1f * y;
    }
    *(f32x4*)(out + obase + (size_t)ix * (NH * NW) + 4 * lane) = v;
  }
}

extern "C" void kernel_launch(void* const* d_in, const int* in_sizes, int n_in,
                              void* d_out, int out_size, void* d_ws, size_t ws_size,
                              hipStream_t stream) {
  (void)in_sizes; (void)n_in; (void)out_size; (void)d_ws; (void)ws_size;
  const float* in1 = (const float*)d_in[0];
  const float* in2 = (const float*)d_in[1];
  float* out = (float*)d_out;
  dim3 grid(NB * NH * PATCH);   // one WG per (batch, row, dy)
  corr_wmma_kernel<<<grid, 128, 0, stream>>>(in1, in2, out);
}
